// PiFormerFFN_12206297055405
// MI455X (gfx1250) — matrix operand order repaired and checked by reading, not compiled, not run on hardware
//
#include <hip/hip_runtime.h>
#include <hip/hip_bf16.h>
#include <stdint.h>

// ---------------- types ----------------
typedef __attribute__((ext_vector_type(16))) __bf16 v16bf;
typedef __attribute__((ext_vector_type(8)))  float  v8f;
typedef __attribute__((ext_vector_type(4)))  int    v4i;
typedef __attribute__((ext_vector_type(4)))  float  v4f;
typedef __attribute__((ext_vector_type(4)))  unsigned short v4us;

#define D_MODEL 2048
#define D_FF    8192
#define NTOK    8192   // BATCH * SEQ = 4 * 2048

// Async global->LDS path (CDNA5). Guarded so host pass / older toolchains
// fall back to the register-staged pipeline.
#if defined(__gfx1250__) && __has_builtin(__builtin_amdgcn_global_load_async_to_lds_b128)
#define ASYNC_LDS 1
#else
#define ASYNC_LDS 0
#endif

#if ASYNC_LDS
#if __has_builtin(__builtin_amdgcn_s_wait_asynccnt)
#define WAIT_ASYNC(n) __builtin_amdgcn_s_wait_asynccnt(n)
#else
#define WAIT_ASYNC(n) asm volatile("s_wait_asynccnt %0" ::"n"(n))
#endif
// builtin takes b128-typed pointers: global (AS1) src, LDS (AS3) dst
typedef __attribute__((address_space(1))) v4i gv4i;
typedef __attribute__((address_space(3))) v4i lv4i;
// one instruction moves 32 lanes x 16B straight into LDS, tracked by ASYNCcnt
__device__ __forceinline__ void async_ld_b128(const void* gsrc, void* ldst) {
    __builtin_amdgcn_global_load_async_to_lds_b128(
        (gv4i*)(uintptr_t)gsrc,
        (lv4i*)(unsigned)(uintptr_t)ldst,
        0, 0);
}
#endif

// f32 -> bf16, round-to-nearest-even
__device__ __forceinline__ unsigned short f2bf(float f) {
    unsigned u = __float_as_uint(f);
    u += 0x7FFFu + ((u >> 16) & 1u);
    return (unsigned short)(u >> 16);
}

// branchless tanh: 1 - 2/(1+exp(2y)); TRANS ops only, no EXEC branching.
__device__ __forceinline__ float fast_tanh(float y) {
    float e = __expf(2.0f * y);
    return 1.0f - __fdividef(2.0f, e + 1.0f);
}

// 16-bit fixed-point quantize (SCALE=0.1) then tanh-approx GELU (jax default)
__device__ __forceinline__ float act_fn(float h) {
    float q = rintf(h * 10.0f);                       // h / SCALE, round-half-even
    q = fminf(fmaxf(q, -32768.0f), 32767.0f);         // 16-bit clamp
    float x = q * 0.1f;
    float x3 = x * x * x;
    float t = fast_tanh(0.7978845608028654f * (x + 0.044715f * x3));
    return 0.5f * x * (1.0f + t);
}

// ---------------- conversion kernels ----------------
__global__ __launch_bounds__(256)
void cvt_f32_bf16(const float* __restrict__ in, unsigned short* __restrict__ out, int n4) {
    int i = blockIdx.x * blockDim.x + threadIdx.x;
    if (i < n4) {
        v4f f = ((const v4f*)in)[i];
        v4us o;
        o.x = f2bf(f.x); o.y = f2bf(f.y); o.z = f2bf(f.z); o.w = f2bf(f.w);
        ((v4us*)out)[i] = o;
    }
}

// W [Nw][Kw] f32 (row-major)  ->  Wt [Kw][Nw] bf16  (B-operand layout)
__global__ __launch_bounds__(256)
void transpose_cvt(const float* __restrict__ W, unsigned short* __restrict__ Wt,
                   int Nw, int Kw) {
    __shared__ float tile[32][33];
    int k0 = blockIdx.x * 32;
    int n0 = blockIdx.y * 32;
    int tx = threadIdx.x, ty = threadIdx.y;   // block (32, 8)
    #pragma unroll
    for (int j = 0; j < 32; j += 8)
        tile[ty + j][tx] = W[(size_t)(n0 + ty + j) * Kw + k0 + tx];
    __syncthreads();
    #pragma unroll
    for (int j = 0; j < 32; j += 8)
        Wt[(size_t)(k0 + ty + j) * Nw + n0 + tx] = f2bf(tile[tx][ty + j]);
}

// ---------------- WMMA GEMM ----------------
// C[M][N] = A[M][K] (bf16, row-major) * B[K][N] (bf16, row-major), f32 accum.
// Block tile 128(M) x 256(N), BK=64. 8 waves, each owns a 64x64 sub-tile
// (4x4 of 16x16 WMMA tiles). Double-buffered LDS fed by async global->LDS
// DMA (ASYNCcnt) so stage k+1 flies during the WMMA work of stage k.
union FragBF {
    v16bf v;
    struct { v4i lo; v4i hi; } s;
};

template <bool ACT>
__global__ __launch_bounds__(256)
void gemm_bf16_wmma(const unsigned short* __restrict__ A,
                    const unsigned short* __restrict__ B,
                    void* __restrict__ Out,
                    int M, int N, int K) {
    constexpr int BM = 128, BN = 256, BK = 64;
    constexpr int LDA = BK + 8;    // 72  (bf16 elems) -> 16B-aligned rows
    constexpr int LDB = BN + 8;    // 264
    __shared__ unsigned short As[2][BM * LDA];   // 2 x 18432 B
    __shared__ unsigned short Bs[2][BK * LDB];   // 2 x 33792 B (~102 KB total)

    const int tid   = threadIdx.x;
    const int lane  = tid & 31;
    const int wave  = tid >> 5;
    const int waveM = wave >> 2;       // 0..1 -> 64-row slab
    const int waveN = wave & 3;        // 0..3 -> 64-col slab
    const int lh    = lane >> 4;       // lane half
    const int ml    = lane & 15;

    const size_t gm = (size_t)blockIdx.y * BM;
    const size_t gn = (size_t)blockIdx.x * BN;

    v8f acc[4][4];
    #pragma unroll
    for (int mt = 0; mt < 4; ++mt)
        #pragma unroll
        for (int nt = 0; nt < 4; ++nt)
            acc[mt][nt] = (v8f)(0.0f);

    // per-thread chunk coordinates (16B chunks)
    // A tile 128x64: 1024 chunks; B tile 64x256: 2048 chunks; 256 threads.
    int arow[4], acol[4], brow[8], bcol[8];
    #pragma unroll
    for (int i = 0; i < 4; ++i) {
        int c = tid + i * 256;
        arow[i] = c >> 3; acol[i] = (c & 7) * 8;
    }
    #pragma unroll
    for (int i = 0; i < 8; ++i) {
        int c = tid + i * 256;
        brow[i] = c >> 5; bcol[i] = (c & 31) * 8;
    }

#if ASYNC_LDS
    // ---- async DMA pipeline: 12 ASYNCcnt ops per stage ----
    {
        #pragma unroll
        for (int i = 0; i < 4; ++i)
            async_ld_b128(&A[(gm + arow[i]) * K + acol[i]], &As[0][arow[i] * LDA + acol[i]]);
        #pragma unroll
        for (int i = 0; i < 8; ++i)
            async_ld_b128(&B[(size_t)brow[i] * N + gn + bcol[i]], &Bs[0][brow[i] * LDB + bcol[i]]);
    }
    int buf = 0;
#else
    // ---- fallback: register-staged pipeline, single LDS buffer ----
    v4i aReg[4], bReg[8];
    #pragma unroll
    for (int i = 0; i < 4; ++i)
        aReg[i] = *(const v4i*)&A[(gm + arow[i]) * K + acol[i]];
    #pragma unroll
    for (int i = 0; i < 8; ++i)
        bReg[i] = *(const v4i*)&B[(size_t)brow[i] * N + gn + bcol[i]];
#endif

    for (int k0 = 0; k0 < K; k0 += BK) {
#if ASYNC_LDS
        if (k0 + BK < K) {
            int kn = k0 + BK, nb = buf ^ 1;
            #pragma unroll
            for (int i = 0; i < 4; ++i)
                async_ld_b128(&A[(gm + arow[i]) * K + kn + acol[i]],
                              &As[nb][arow[i] * LDA + acol[i]]);
            #pragma unroll
            for (int i = 0; i < 8; ++i)
                async_ld_b128(&B[(size_t)(kn + brow[i]) * N + gn + bcol[i]],
                              &Bs[nb][brow[i] * LDB + bcol[i]]);
            WAIT_ASYNC(12);   // in-order: the 12 older ops (this stage) are done
        } else {
            WAIT_ASYNC(0);
        }
        __syncthreads();      // stage visible to all waves
        const unsigned short* Asb = As[buf];
        const unsigned short* Bsb = Bs[buf];
#else
        __syncthreads();      // previous stage's readers done
        #pragma unroll
        for (int i = 0; i < 4; ++i)
            *(v4i*)&As[0][arow[i] * LDA + acol[i]] = aReg[i];
        #pragma unroll
        for (int i = 0; i < 8; ++i)
            *(v4i*)&Bs[0][brow[i] * LDB + bcol[i]] = bReg[i];
        __syncthreads();
        if (k0 + BK < K) {
            int kn = k0 + BK;
            #pragma unroll
            for (int i = 0; i < 4; ++i)
                aReg[i] = *(const v4i*)&A[(gm + arow[i]) * K + kn + acol[i]];
            #pragma unroll
            for (int i = 0; i < 8; ++i)
                bReg[i] = *(const v4i*)&B[(size_t)(kn + brow[i]) * N + gn + bcol[i]];
        }
        const unsigned short* Asb = As[0];
        const unsigned short* Bsb = Bs[0];
#endif

        // compute: 2 K-steps of 32, 16 WMMAs each
        #pragma unroll
        for (int kk = 0; kk < BK; kk += 32) {
            FragBF b[4];
            // B fragment: lane holds column N=ml, K {kk+8*lh..+7},{kk+16+8*lh..+7}
            #pragma unroll
            for (int nt = 0; nt < 4; ++nt) {
                const unsigned short* bp =
                    &Bsb[(kk + lh * 8) * LDB + waveN * 64 + nt * 16 + ml];
                #pragma unroll
                for (int s = 0; s < 16; ++s)
                    b[nt].v[s] = __builtin_bit_cast(__bf16, bp[(s + (s >> 3) * 8) * LDB]);
            }
            #pragma unroll
            for (int mt = 0; mt < 4; ++mt) {
                // A fragment: lane<16 -> M=ml, K {kk..+7},{kk+16..+23};
                //             lane>=16 -> K {kk+8..+15},{kk+24..+31}
                FragBF a;
                const unsigned short* ap =
                    &Asb[(waveM * 64 + mt * 16 + ml) * LDA + kk + lh * 8];
                a.s.lo = *(const v4i*)ap;
                a.s.hi = *(const v4i*)(ap + 16);
                #pragma unroll
                for (int nt = 0; nt < 4; ++nt)
                    acc[mt][nt] = __builtin_amdgcn_wmma_f32_16x16x32_bf16(
                        false, a.v, false, b[nt].v,
                        (short)0, acc[mt][nt], false, false);
            }
        }
        __syncthreads();      // all waves done reading this buffer
#if ASYNC_LDS
        buf ^= 1;
#endif
    }

    // --- epilogue: C layout = row m0 + (lane>=16 ? 8 : 0) + i, col n0 + (lane&15)
    #pragma unroll
    for (int mt = 0; mt < 4; ++mt) {
        size_t row0 = gm + waveM * 64 + mt * 16 + lh * 8;
        #pragma unroll
        for (int nt = 0; nt < 4; ++nt) {
            size_t col = gn + waveN * 64 + nt * 16 + ml;
            #pragma unroll
            for (int i = 0; i < 8; ++i) {
                float v = acc[mt][nt][i];
                if (ACT) {
                    ((unsigned short*)Out)[(row0 + i) * (size_t)N + col] = f2bf(act_fn(v));
                } else {
                    ((float*)Out)[(row0 + i) * (size_t)N + col] = v;
                }
            }
        }
    }
}

// ---------------- launcher ----------------
extern "C" void kernel_launch(void* const* d_in, const int* in_sizes, int n_in,
                              void* d_out, int out_size, void* d_ws, size_t ws_size,
                              hipStream_t stream) {
    const float* x  = (const float*)d_in[0];   // [4,2048,2048]
    const float* W1 = (const float*)d_in[1];   // [8192,2048]
    const float* W2 = (const float*)d_in[2];   // [2048,8192]
    float* out = (float*)d_out;                // [4,2048,2048] f32

    // workspace layout (bf16 = unsigned short)
    unsigned short* xb  = (unsigned short*)d_ws;                 // [NTOK][D_MODEL]
    unsigned short* w1t = xb  + (size_t)NTOK * D_MODEL;          // [D_MODEL][D_FF]
    unsigned short* w2t = w1t + (size_t)D_MODEL * D_FF;          // [D_FF][D_MODEL]
    unsigned short* ab  = w2t + (size_t)D_FF * D_MODEL;          // [NTOK][D_FF]

    // 1) x -> bf16
    {
        int n4 = (NTOK * D_MODEL) / 4;
        cvt_f32_bf16<<<(n4 + 255) / 256, 256, 0, stream>>>(x, xb, n4);
    }
    // 2) W1 [D_FF][D_MODEL] -> w1t [D_MODEL][D_FF] bf16
    transpose_cvt<<<dim3(D_MODEL / 32, D_FF / 32), dim3(32, 8), 0, stream>>>(
        W1, w1t, D_FF, D_MODEL);
    // 3) W2 [D_MODEL][D_FF] -> w2t [D_FF][D_MODEL] bf16
    transpose_cvt<<<dim3(D_FF / 32, D_MODEL / 32), dim3(32, 8), 0, stream>>>(
        W2, w2t, D_MODEL, D_FF);

    // 4) GEMM1 + activation: ab = act(xb @ w1t), M=NTOK, N=D_FF, K=D_MODEL
    gemm_bf16_wmma<true><<<dim3(D_FF / 256, NTOK / 128), 256, 0, stream>>>(
        xb, w1t, (void*)ab, NTOK, D_FF, D_MODEL);

    // 5) GEMM2: out = ab @ w2t, M=NTOK, N=D_MODEL, K=D_FF
    gemm_bf16_wmma<false><<<dim3(D_MODEL / 256, NTOK / 128), 256, 0, stream>>>(
        ab, w2t, (void*)out, NTOK, D_MODEL, D_FF);
}
